// AttentionPairBias_86019605004395
// MI455X (gfx1250) — matrix operand order repaired and checked by reading, not compiled, not run on hardware
//
#include <hip/hip_runtime.h>
#include <hip/hip_bf16.h>
#include <math.h>

#define L 1024
#define CS 256
#define CZ 128
#define H 8
#define D 32
#define EPS 1e-5f
#define QSCALE 0.17677669529663687f  // 1/sqrt(D)

typedef __attribute__((ext_vector_type(16))) _Float16 v16h;
typedef __attribute__((ext_vector_type(8)))  _Float16 v8h;
typedef __attribute__((ext_vector_type(8)))  float    v8f;

// ---------------------------------------------------------------- helpers
__device__ __forceinline__ float wave_sum32(float v) {
#pragma unroll
    for (int o = 16; o > 0; o >>= 1) v += __shfl_xor(v, o, 32);
    return v;
}

// ---------------------------------------------------------------- K0: fold g_z into Wb, precompute A_h, C_h
__global__ void prep_wb(const float* __restrict__ gz, const float* __restrict__ bz,
                        const float* __restrict__ Wb, float* __restrict__ Wgb,
                        float* __restrict__ AC) {
    int c = threadIdx.x;  // 128 threads
    if (c < CZ) {
        float g = gz[c];
#pragma unroll
        for (int h = 0; h < H; ++h) Wgb[c * H + h] = g * Wb[c * H + h];
    }
    if (threadIdx.x < H) {
        int h = threadIdx.x;
        float a = 0.f, cc = 0.f;
        for (int k = 0; k < CZ; ++k) {
            a  += gz[k] * Wb[k * H + h];
            cc += bz[k] * Wb[k * H + h];
        }
        AC[h]     = a;
        AC[H + h] = cc;
    }
}

// ---------------------------------------------------------------- K1: layernorm of s  [1024 x 256]
__global__ void ln_s(const float* __restrict__ s, const float* __restrict__ g,
                     const float* __restrict__ b, float* __restrict__ sln) {
    __shared__ float red[8];
    __shared__ float mv[2];
    int row = blockIdx.x, t = threadIdx.x;
    float x = s[(size_t)row * CS + t];

    float sum = wave_sum32(x);
    if ((t & 31) == 0) red[t >> 5] = sum;
    __syncthreads();
    if (t == 0) {
        float v = 0.f;
        for (int i = 0; i < 8; ++i) v += red[i];
        mv[0] = v * (1.0f / CS);
    }
    __syncthreads();
    float m = mv[0];
    float d = x - m;
    float sq = wave_sum32(d * d);
    __syncthreads();
    if ((t & 31) == 0) red[t >> 5] = sq;
    __syncthreads();
    if (t == 0) {
        float v = 0.f;
        for (int i = 0; i < 8; ++i) v += red[i];
        mv[1] = rsqrtf(v * (1.0f / CS) + EPS);
    }
    __syncthreads();
    sln[(size_t)row * CS + t] = d * mv[1] * g[t] + b[t];
}

// ---------------------------------------------------------------- K2: Q/K/V/G projections (row per block)
// Q,K stored head-major [H][L][D] f16 (Q pre-scaled by 1/sqrt(D)).
// V stored TRANSPOSED  [H][D][L] f16 so attn's V B-fragments are contiguous.
__global__ void proj(const float* __restrict__ sln, const float* __restrict__ Wq,
                     const float* __restrict__ bq, const float* __restrict__ Wk,
                     const float* __restrict__ Wv, const float* __restrict__ Wg,
                     _Float16* __restrict__ Qh, _Float16* __restrict__ Kh,
                     _Float16* __restrict__ Vt, float* __restrict__ G) {
    __shared__ float row[CS];
    int i = blockIdx.x, t = threadIdx.x;
    row[t] = sln[(size_t)i * CS + t];
    __syncthreads();
    float q = 0.f, k = 0.f, v = 0.f, gg = 0.f;
    for (int c = 0; c < CS; ++c) {
        float x = row[c];
        q  += x * Wq[c * CS + t];
        k  += x * Wk[c * CS + t];
        v  += x * Wv[c * CS + t];
        gg += x * Wg[c * CS + t];
    }
    q += bq[t];
    int h = t >> 5, d = t & 31;
    size_t idx = ((size_t)h * L + i) * D + d;   // [H][L][D]
    Qh[idx] = (_Float16)(q * QSCALE);
    Kh[idx] = (_Float16)k;
    Vt[((size_t)h * D + d) * L + i] = (_Float16)v;   // [H][D][L]
    G[(size_t)i * CS + t] = 1.0f / (1.0f + __expf(-gg));
}

// ---------------------------------------------------------------- K3: fused LN(z)+Wb -> head-major bias Bh[H][L][L]
// 4 lanes per pair, 8 pairs per wave, 64 pairs (one j-strip) per 256-thread
// block. This is the 512MB roofline kernel: ~1 b128 vmem + ~32 FMA-instr +
// ~8 LDS-b128 + ~2.5 shuffles per pair, outputs staged in LDS and written
// as dense 256B bursts per head.
__global__ void pair_bias(const float* __restrict__ z, const float* __restrict__ Wgb,
                          const float* __restrict__ AC, float* __restrict__ Bh) {
    __shared__ float lW[CZ * H];   // 4 KB
    __shared__ float lAC[16];
    __shared__ float sB[H][64];    // 2 KB staging for coalesced writes
    int t = threadIdx.x;
    for (int k = t; k < CZ * H; k += 256) lW[k] = Wgb[k];
    if (t < 16) lAC[t] = AC[t];
    __syncthreads();

    int wave = t >> 5, lane = t & 31;
    int ps = lane >> 2, q = lane & 3;          // pair-in-wave, quad-in-pair
    int i  = blockIdx.x >> 4;                  // 16 blocks per i-row
    int j0 = (blockIdx.x & 15) * 64;
    int jj = wave * 8 + ps;                    // 0..63
    size_t pair = (size_t)i * L + j0 + jj;
    const float* zp = z + pair * CZ + q * 4;   // quad q owns c = k*16 + 4q + e

    float4 zv[8];
#pragma unroll
    for (int k = 0; k < 8; ++k) zv[k] = *(const float4*)(zp + k * 16);

    float sum = 0.f, sq = 0.f;
#pragma unroll
    for (int k = 0; k < 8; ++k) {
        sum += zv[k].x + zv[k].y + zv[k].z + zv[k].w;
        sq  += zv[k].x * zv[k].x + zv[k].y * zv[k].y +
               zv[k].z * zv[k].z + zv[k].w * zv[k].w;
    }
    sum += __shfl_xor(sum, 1, 32); sum += __shfl_xor(sum, 2, 32);
    sq  += __shfl_xor(sq, 1, 32);  sq  += __shfl_xor(sq, 2, 32);
    float m = sum * (1.0f / CZ);
    float r = rsqrtf(sq * (1.0f / CZ) - m * m + EPS);

    float acc[H];
#pragma unroll
    for (int h = 0; h < H; ++h) acc[h] = 0.f;
#pragma unroll
    for (int k = 0; k < 8; ++k) {
        float vs[4] = {zv[k].x, zv[k].y, zv[k].z, zv[k].w};
#pragma unroll
        for (int e = 0; e < 4; ++e) {
            const float* w = &lW[(k * 16 + q * 4 + e) * H];  // 32B contiguous
            float zval = vs[e];
#pragma unroll
            for (int h = 0; h < H; ++h) acc[h] += zval * w[h];
        }
    }
#pragma unroll
    for (int h = 0; h < H; ++h) {
        acc[h] += __shfl_xor(acc[h], 1, 32);
        acc[h] += __shfl_xor(acc[h], 2, 32);
    }
    // lane with quad q owns heads 2q, 2q+1
#pragma unroll
    for (int h = 0; h < H; ++h) {
        if ((h >> 1) == q)
            sB[h][jj] = r * (acc[h] - m * lAC[h]) + lAC[H + h];
    }
    __syncthreads();
    // dense write-out: 64 consecutive floats per head per block
#pragma unroll
    for (int rep = 0; rep < 2; ++rep) {
        int lin = t + rep * 256;               // 0..511
        int h = lin >> 6, jc = lin & 63;
        Bh[(size_t)h * L * L + (size_t)i * L + j0 + jc] = sB[h][jc];
    }
}

// ---------------------------------------------------------------- K4: WMMA flash attention
// grid (L/16, H), block = one wave32. D=32 == one v_wmma_f32_16x16x32_f16 K-dim.
__global__ void __launch_bounds__(32)
attn(const _Float16* __restrict__ Qh, const _Float16* __restrict__ Kh,
     const _Float16* __restrict__ Vt, const float* __restrict__ Bh,
     const float* __restrict__ G, const unsigned char* __restrict__ mask,
     float* __restrict__ AO) {
    __shared__ _Float16 ldsP[16 * 32];   // P tile, C-layout -> A-layout staging

    int i0 = blockIdx.x * 16;
    int h  = blockIdx.y;
    int lane = threadIdx.x;
    int n  = lane & 15;
    int hv = lane >> 4;

    const _Float16* Qb  = Qh + ((size_t)h * L + i0) * D;
    const _Float16* Kb  = Kh + (size_t)h * L * D;
    const _Float16* Vb  = Vt + (size_t)h * D * L;        // [D][L]
    const float*    Bb  = Bh + (size_t)h * L * L;        // [L][L]

    // A fragment of Q [16 x 32]: one contiguous 32B chunk per lane
    // (K = hv*16..hv*16+15)
    v16h aq;
    {
        aq = *(const v16h*)(Qb + (size_t)n * D + hv * 16);
    }

    v8f o0 = {}, o1 = {};
    float mrow[8], lrow[8];
#pragma unroll
    for (int r = 0; r < 8; ++r) { mrow[r] = -1e30f; lrow[r] = 0.f; }

    for (int j0 = 0; j0 < L; j0 += 32) {
        // prefetch next chunk of K and V (global_prefetch_b8)
        __builtin_prefetch(Kb + (size_t)(j0 + 32 + n) * D, 0, 0);
        __builtin_prefetch(Vb + (size_t)n * L + j0 + 32, 0, 0);

        // ---- S = Q @ K^T, two 16-wide j subtiles (one WMMA each, K=D=32).
        // B-fragment per lane = 32 contiguous bytes of K row (j0+n):
        // lanes 0-15 hold K=0-15, lanes 16-31 hold K=16-31.
        v16h bk0 = *(const v16h*)(Kb + (size_t)(j0 + n) * D + hv * 16);
        v16h bk1 = *(const v16h*)(Kb + (size_t)(j0 + 16 + n) * D + hv * 16);
        v8f s0 = {}, s1 = {};
        s0 = __builtin_amdgcn_wmma_f32_16x16x32_f16(false, aq, false, bk0,
                                                    (short)0, s0, false, false);
        s1 = __builtin_amdgcn_wmma_f32_16x16x32_f16(false, aq, false, bk1,
                                                    (short)0, s1, false, false);

        int j_a = j0 + n, j_b = j0 + 16 + n;          // this lane's columns
        float ma = (mask[j_a] != 0) ? 0.f : 1.f;
        float mb = (mask[j_b] != 0) ? 0.f : 1.f;

        // ---- online softmax per row (row r+8*hv lives in this 16-lane group)
#pragma unroll
        for (int r = 0; r < 8; ++r) {
            int row = r + 8 * hv;
            size_t base = (size_t)(i0 + row) * L;     // dense 64B row reads
            float sa = s0[r] + Bb[base + j_a];
            float sb = s1[r] + Bb[base + j_b];
            sa = (ma == 0.f) ? sa : -1e30f;           // v_cndmask, no branch
            sb = (mb == 0.f) ? sb : -1e30f;

            float tmx = fmaxf(sa, sb);
#pragma unroll
            for (int o = 8; o > 0; o >>= 1) tmx = fmaxf(tmx, __shfl_xor(tmx, o, 32));
            float mnew  = fmaxf(mrow[r], tmx);
            float alpha = __expf(mrow[r] - mnew);
            float pa = (sa < -5e29f) ? 0.f : __expf(sa - mnew);
            float pb = (sb < -5e29f) ? 0.f : __expf(sb - mnew);
            float ps = pa + pb;
#pragma unroll
            for (int o = 8; o > 0; o >>= 1) ps += __shfl_xor(ps, o, 32);
            lrow[r] = lrow[r] * alpha + ps;
            mrow[r] = mnew;
            o0[r] *= alpha;
            o1[r] *= alpha;
            ldsP[row * 32 + n]      = (_Float16)pa;
            ldsP[row * 32 + 16 + n] = (_Float16)pb;
        }
        asm volatile("s_wait_dscnt 0" ::: "memory");  // wave-local LDS RAW fence

        // ---- O += P @ V (A = P from LDS, V B-frags contiguous from Vt)
        v16h ap;
        {
            ap = *(const v16h*)(&ldsP[n * 32 + hv * 16]);
        }
        v16h bv0 = *(const v16h*)(Vb + (size_t)n * L + j0 + hv * 16);
        v16h bv1 = *(const v16h*)(Vb + (size_t)(16 + n) * L + j0 + hv * 16);
        o0 = __builtin_amdgcn_wmma_f32_16x16x32_f16(false, ap, false, bv0,
                                                    (short)0, o0, false, false);
        o1 = __builtin_amdgcn_wmma_f32_16x16x32_f16(false, ap, false, bv1,
                                                    (short)0, o1, false, false);
    }

    // ---- epilogue: 1/lsum (0 if fully masked), gate by G, write [L,256]
#pragma unroll
    for (int r = 0; r < 8; ++r) {
        int row = r + 8 * hv;
        int i = i0 + row;
        float inv = (lrow[r] > 0.f) ? 1.0f / lrow[r] : 0.f;
        float g0 = G[(size_t)i * CS + h * D + n];
        float g1 = G[(size_t)i * CS + h * D + 16 + n];
        AO[(size_t)i * CS + h * D + n]      = o0[r] * inv * g0;
        AO[(size_t)i * CS + h * D + 16 + n] = o1[r] * inv * g1;
    }
}

// ---------------------------------------------------------------- K5: output projection + row mask
__global__ void final_proj(const float* __restrict__ AO, const float* __restrict__ Wo,
                           const unsigned char* __restrict__ mask, float* __restrict__ out) {
    __shared__ float row[CS];
    int i = blockIdx.x, t = threadIdx.x;
    row[t] = AO[(size_t)i * CS + t];
    __syncthreads();
    float acc = 0.f;
    for (int c = 0; c < CS; ++c) acc += row[c] * Wo[c * CS + t];
    out[(size_t)i * CS + t] = mask[i] ? acc : 0.f;
}

// ---------------------------------------------------------------- launcher
extern "C" void kernel_launch(void* const* d_in, const int* in_sizes, int n_in,
                              void* d_out, int out_size, void* d_ws, size_t ws_size,
                              hipStream_t stream) {
    const float* s    = (const float*)d_in[0];
    const float* z    = (const float*)d_in[1];
    const unsigned char* res_mask = (const unsigned char*)d_in[2];
    const float* g_s  = (const float*)d_in[3];
    const float* b_s  = (const float*)d_in[4];
    const float* g_z  = (const float*)d_in[5];
    const float* b_z  = (const float*)d_in[6];
    const float* Wq   = (const float*)d_in[7];
    const float* bq   = (const float*)d_in[8];
    const float* Wk   = (const float*)d_in[9];
    const float* Wv   = (const float*)d_in[10];
    const float* Wb   = (const float*)d_in[11];
    const float* Wg   = (const float*)d_in[12];
    const float* Wo   = (const float*)d_in[13];
    float* out = (float*)d_out;

    // workspace carve-up (all chunks 256B-aligned)
    char* w = (char*)d_ws;
    float*    sln = (float*)w;    w += (size_t)L * CS * 4;       // 1 MB
    float*    G   = (float*)w;    w += (size_t)L * CS * 4;       // 1 MB
    float*    AO  = (float*)w;    w += (size_t)L * CS * 4;       // 1 MB
    _Float16* Qh  = (_Float16*)w; w += (size_t)H * L * D * 2;    // 512 KB
    _Float16* Kh  = (_Float16*)w; w += (size_t)H * L * D * 2;    // 512 KB
    _Float16* Vt  = (_Float16*)w; w += (size_t)H * L * D * 2;    // 512 KB
    float*    Wgb = (float*)w;    w += (size_t)CZ * H * 4;       // 4 KB
    float*    AC  = (float*)w;    w += 256;                      // 64 B used
    float*    Bh  = (float*)w;    /* H*L*L*4 = 32 MB, head-major */

    prep_wb   <<<1, 128, 0, stream>>>(g_z, b_z, Wb, Wgb, AC);
    ln_s      <<<L, CS, 0, stream>>>(s, g_s, b_s, sln);
    proj      <<<L, CS, 0, stream>>>(sln, Wq, bq, Wk, Wv, Wg, Qh, Kh, Vt, G);
    pair_bias <<<L * 16, 256, 0, stream>>>(z, Wgb, AC, Bh);
    attn      <<<dim3(L / 16, H), 32, 0, stream>>>(Qh, Kh, Vt, Bh, G, res_mask, AO);
    final_proj<<<L, CS, 0, stream>>>(AO, Wo, res_mask, out);
}
